// TreeReWeightedMessagePassing_12395275616825
// MI455X (gfx1250) — 8-bit matrix operand order repaired and checked by reading, compiled, not run on hardware
//
#include <hip/hip_runtime.h>
#include <cstdint>
#include <cstddef>

#define NN 2048
#define TT 4
#define EE 4096
#define MAXIT 10

typedef int v8i __attribute__((ext_vector_type(8)));

__device__ __forceinline__ float lse2(float a, float b) {
    float m = fmaxf(a, b);
    return m + logf(expf(a - m) + expf(b - m));
}

__global__ void k_setup(const float* __restrict__ J, const int* __restrict__ msg_node,
                        int* __restrict__ eins, int* __restrict__ eouts,
                        float* __restrict__ je, float* __restrict__ lm,
                        int8_t* __restrict__ lm8) {
    int tid = blockIdx.x * blockDim.x + threadIdx.x;
    if (tid >= TT * EE) return;
    int t = tid / EE, e = tid % EE;
    int ein  = msg_node[tid * 2 + 0];
    int eout = msg_node[tid * 2 + 1];
    eins[tid]  = ein;
    eouts[tid] = eout;
    je[tid] = J[(size_t)ein * NN + eout];
    lm[tid * 2 + 0] = -0.6931471805599453f;
    lm[tid * 2 + 1] = -0.6931471805599453f;
    lm8[(size_t)t * 2 * EE + 0 * EE + e] = 0;
    lm8[(size_t)t * 2 * EE + 1 * EE + e] = 0;
}

__global__ void k_pack(const int* __restrict__ madj, uint8_t* __restrict__ packed) {
    int tid = blockIdx.x * blockDim.x + threadIdx.x;
    if (tid >= TT * EE * (EE / 4)) return;
    int n  = tid % EE;
    int r  = tid / EE;
    int k4 = r % (EE / 4);
    int t  = r / (EE / 4);
    int kb = k4 * 4;
    const int* src = madj + ((size_t)t * EE + kb) * EE + n;
    unsigned dw = 0;
#pragma unroll
    for (int j = 0; j < 4; ++j)
        dw |= (unsigned)(src[(size_t)j * EE] & 0xFF) << (8 * j);
    int ntile = n >> 4;
    int kblk  = kb >> 6;
    int kk    = kb & 63;
    int g = kk >> 5, rem = kk & 31;
    int h = rem >> 4;
    int v = (rem & 15) >> 2;
    int d = g * 4 + v;
    int L = h * 16 + (n & 15);
    size_t frag = ((size_t)(t * (EE / 16) + ntile) * (EE / 64) + kblk) * 1024;
    *(unsigned*)(packed + frag + (size_t)(L * 8 + d) * 4) = dw;
}

__global__ void __launch_bounds__(256) k_agg(const uint8_t* __restrict__ packed,
                                             const int* __restrict__ madj,
                                             const int8_t* __restrict__ lm8,
                                             int* __restrict__ agg, int packedFlag) {
    __shared__ int8_t slm[3 * EE];
    int tix = threadIdx.x;
    int t = blockIdx.x >> 5;

    *(int4*)(slm + 2 * EE + tix * 16) = make_int4(0, 0, 0, 0);
    {
        unsigned ldso = (unsigned)(uintptr_t)(slm) + (unsigned)(tix * 32);
        const int8_t* g = lm8 + (size_t)t * 2 * EE + (size_t)tix * 32;
        asm volatile("global_load_async_to_lds_b128 %0, %1, off\n\t"
                     "global_load_async_to_lds_b128 %0, %1, off offset:16"
                     :: "v"(ldso), "v"(g) : "memory");
        asm volatile("s_wait_asynccnt 0" ::: "memory");
    }
    __syncthreads();

    int wid = blockIdx.x * 8 + (tix >> 5);
    int L = tix & 31;
    int ntile = wid & 255;
    int n0 = ntile << 4;
    int m = L & 15, h = L >> 4;
    const int8_t* As = slm + (size_t)((m < 2) ? m : 2) * EE;
    v8i acc = {};

    if (packedFlag) {
        const uint8_t* bp = packed +
            ((size_t)(t * (EE / 16) + ntile) * (EE / 64)) * 1024 + (size_t)L * 32;
#pragma unroll 2
        for (int kblk = 0; kblk < EE / 64; ++kblk) {
            v8i afr, bfr;
#pragma unroll
            for (int p = 0; p < 4; ++p) {
                unsigned long long a2 =
                    *(const unsigned long long*)(As + kblk * 64 + (p >> 1) * 32 + h * 16 + (p & 1) * 8);
                afr[2 * p + 0] = (int)(unsigned)a2;
                afr[2 * p + 1] = (int)(a2 >> 32);
            }
            const int4* fp = (const int4*)(bp + (size_t)kblk * 1024);
            int4 q0 = fp[0], q1 = fp[1];
            bfr[0] = q0.x; bfr[1] = q0.y; bfr[2] = q0.z; bfr[3] = q0.w;
            bfr[4] = q1.x; bfr[5] = q1.y; bfr[6] = q1.z; bfr[7] = q1.w;
            acc = __builtin_amdgcn_wmma_i32_16x16x64_iu8(true, afr, false, bfr, acc,
                                                         false, false);
        }
    } else {
        int n = n0 + m;
        const int* srcT = madj + (size_t)t * EE * EE;
        for (int kblk = 0; kblk < EE / 64; ++kblk) {
            v8i afr, bfr;
#pragma unroll
            for (int p = 0; p < 4; ++p) {
                unsigned long long a2 =
                    *(const unsigned long long*)(As + kblk * 64 + (p >> 1) * 32 + h * 16 + (p & 1) * 8);
                afr[2 * p + 0] = (int)(unsigned)a2;
                afr[2 * p + 1] = (int)(a2 >> 32);
            }
#pragma unroll
            for (int d = 0; d < 8; ++d) {
                int g = d >> 2, v = d & 3;
                int kb = kblk * 64 + g * 32 + h * 16 + v * 4;
                unsigned dw = 0;
#pragma unroll
                for (int j = 0; j < 4; ++j)
                    dw |= (unsigned)(srcT[(size_t)(kb + j) * EE + n] & 0xFF) << (8 * j);
                bfr[d] = (int)dw;
            }
            acc = __builtin_amdgcn_wmma_i32_16x16x64_iu8(true, afr, false, bfr, acc,
                                                         false, false);
        }
    }

    if (L < 16) {
        agg[((size_t)t * EE + n0 + L) * 2 + 0] = acc[0];
        agg[((size_t)t * EE + n0 + L) * 2 + 1] = acc[1];
    }
}

__global__ void k_upd(const float* __restrict__ bvec, const float* __restrict__ mask,
                      const int* __restrict__ eins, const float* __restrict__ je,
                      const int* __restrict__ agg, float* __restrict__ lm,
                      int8_t* __restrict__ lm8) {
    int tid = blockIdx.x * blockDim.x + threadIdx.x;
    if (tid >= TT * EE) return;
    int t = tid / EE, e = tid % EE;
    float Je = je[tid];
    float bv = bvec[eins[tid]];
    float a0 = (float)agg[tid * 2 + 0];
    float a1 = (float)agg[tid * 2 + 1];
    const float* mk = mask + (size_t)e * 4;
    float t00 = mk[0] * ( bv + Je + a0);
    float t01 = mk[1] * ( bv - Je + a0);
    float t10 = mk[2] * (-bv - Je + a1);
    float t11 = mk[3] * (-bv + Je + a1);
    float lm0 = lm[tid * 2 + 0], lm1 = lm[tid * 2 + 1];
    float n0v = 0.5f * lse2(t00, t10) + 0.5f * lm0;
    float n1v = 0.5f * lse2(t01, t11) + 0.5f * lm1;
    float z = lse2(n0v, n1v);
    n0v -= z; n1v -= z;
    lm[tid * 2 + 0] = n0v;
    lm[tid * 2 + 1] = n1v;
    int q0 = (int)n0v; q0 = max(-128, min(127, q0));
    int q1 = (int)n1v; q1 = max(-128, min(127, q1));
    lm8[(size_t)t * 2 * EE + 0 * EE + e] = (int8_t)q0;
    lm8[(size_t)t * 2 * EE + 1 * EE + e] = (int8_t)q1;
}

__global__ void k_segsum(const float* __restrict__ bvec, const int* __restrict__ eouts,
                         const float* __restrict__ lm, float* __restrict__ lpt) {
    int tid = blockIdx.x * blockDim.x + threadIdx.x;
    if (tid >= TT * NN) return;
    int t = tid / NN, n = tid % NN;
    const int*   eo  = eouts + (size_t)t * EE;
    const float* lmt = lm + (size_t)t * EE * 2;
    float s0 = 0.f, s1 = 0.f;
    for (int e = 0; e < EE; ++e) {
        bool hit = (eo[e] == n);
        s0 += hit ? lmt[e * 2 + 0] : 0.f;
        s1 += hit ? lmt[e * 2 + 1] : 0.f;
    }
    float l0 =  bvec[n] + s0;
    float l1 = -bvec[n] + s1;
    float z = lse2(l0, l1);
    lpt[tid * 2 + 0] = l0 - z;
    lpt[tid * 2 + 1] = l1 - z;
}

__global__ void k_out(const float* __restrict__ lpt, float* __restrict__ out) {
    int n = blockIdx.x * blockDim.x + threadIdx.x;
    if (n >= NN) return;
    float c0 = 0.f, c1 = 0.f;
#pragma unroll
    for (int t = 0; t < TT; ++t) {
        c0 += lpt[((size_t)t * NN + n) * 2 + 0];
        c1 += lpt[((size_t)t * NN + n) * 2 + 1];
        out[2 * NN + n * TT + t] = expf(c0 * 0.25f);
    }
    out[n * 2 + 0] = expf(c0 * 0.25f);
    out[n * 2 + 1] = expf(c1 * 0.25f);
}

extern "C" void kernel_launch(void* const* d_in, const int* in_sizes, int n_in,
                              void* d_out, int out_size, void* d_ws, size_t ws_size,
                              hipStream_t stream) {
    const float* J        = (const float*)d_in[0];
    const float* b        = (const float*)d_in[1];
    const int*   msg_node = (const int*)d_in[2];
    const int*   msg_adj  = (const int*)d_in[3];
    const float* mask     = (const float*)d_in[4];
    float* out = (float*)d_out;
    uint8_t* ws = (uint8_t*)d_ws;

    size_t off = 0;
    int*     eins  = (int*)(ws + off);    off += (size_t)TT * EE * 4;
    int*     eouts = (int*)(ws + off);    off += (size_t)TT * EE * 4;
    float*   je    = (float*)(ws + off);  off += (size_t)TT * EE * 4;
    float*   lm    = (float*)(ws + off);  off += (size_t)TT * EE * 2 * 4;
    int*     agg   = (int*)(ws + off);    off += (size_t)TT * EE * 2 * 4;
    int8_t*  lm8   = (int8_t*)(ws + off); off += (size_t)TT * 2 * EE;
    float*   lpt   = (float*)(ws + off);  off += (size_t)TT * NN * 2 * 4;
    const size_t packOff = (size_t)1 << 20;
    uint8_t* packed = ws + packOff;
    int packedFlag = (ws_size >= packOff + (size_t)TT * EE * EE) ? 1 : 0;

    k_setup<<<(TT * EE) / 256, 256, 0, stream>>>(J, msg_node, eins, eouts, je, lm, lm8);
    if (packedFlag)
        k_pack<<<(TT * EE * (EE / 4)) / 256, 256, 0, stream>>>(msg_adj, packed);

    for (int it = 0; it < MAXIT; ++it) {
        k_agg<<<(TT * (EE / 16)) / 8, 256, 0, stream>>>(packed, msg_adj, lm8, agg,
                                                        packedFlag);
        k_upd<<<(TT * EE) / 256, 256, 0, stream>>>(b, mask, eins, je, agg, lm, lm8);
    }

    k_segsum<<<(TT * NN) / 256, 256, 0, stream>>>(b, eouts, lm, lpt);
    k_out<<<NN / 256, 256, 0, stream>>>(lpt, out);
}
